// Block_58729382805721
// MI455X (gfx1250) — matrix operand order repaired and checked by reading, not compiled, not run on hardware
//
#include <hip/hip_runtime.h>
#include <math.h>

// ---------------------------------------------------------------------------
// Transformer block for MI455X (gfx1250, wave32, WMMA 16x16x32 bf16).
// B=4, T=1024, C=1024, H=16, d=64.  All GEMMs via v_wmma_f32_16x16x32_bf16.
// ---------------------------------------------------------------------------

typedef __attribute__((ext_vector_type(16))) __bf16 v16bf;
typedef __attribute__((ext_vector_type(8)))  float  v8f;

#define INFF __builtin_inff()

// ---------------------------------------------------------------------------
// Weight transpose + fp32->bf16:  in[K][N] (row-major) -> out[N][K]
// ---------------------------------------------------------------------------
__global__ __launch_bounds__(256) void transpose_to_bf16(
    const float* __restrict__ in, __bf16* __restrict__ out, int K, int N)
{
    __shared__ float sh[32][33];
    int kb = blockIdx.x * 32, nb = blockIdx.y * 32;
    for (int i = threadIdx.y; i < 32; i += 8)
        sh[i][threadIdx.x] = in[(size_t)(kb + i) * N + nb + threadIdx.x];
    __syncthreads();
    for (int i = threadIdx.y; i < 32; i += 8)
        out[(size_t)(nb + i) * K + kb + threadIdx.x] = (__bf16)sh[threadIdx.x][i];
}

// ---------------------------------------------------------------------------
// LayerNorm: one 256-thread block per row of 1024 fp32, output bf16
// ---------------------------------------------------------------------------
__global__ __launch_bounds__(256) void layernorm_bf16(
    const float* __restrict__ x, const float* __restrict__ g,
    const float* __restrict__ be, __bf16* __restrict__ y)
{
    int row = blockIdx.x;
    const float4 v = ((const float4*)(x + (size_t)row * 1024))[threadIdx.x];
    float s1 = v.x + v.y + v.z + v.w;
    float s2 = v.x * v.x + v.y * v.y + v.z * v.z + v.w * v.w;
    for (int o = 16; o > 0; o >>= 1) {
        s1 += __shfl_xor(s1, o, 32);
        s2 += __shfl_xor(s2, o, 32);
    }
    __shared__ float r1[8], r2[8];
    int w = threadIdx.x >> 5, lane = threadIdx.x & 31;
    if (lane == 0) { r1[w] = s1; r2[w] = s2; }
    __syncthreads();
    if (w == 0) {
        s1 = (lane < 8) ? r1[lane] : 0.f;
        s2 = (lane < 8) ? r2[lane] : 0.f;
        for (int o = 4; o > 0; o >>= 1) {
            s1 += __shfl_xor(s1, o, 32);
            s2 += __shfl_xor(s2, o, 32);
        }
        if (lane == 0) { r1[0] = s1; r2[0] = s2; }
    }
    __syncthreads();
    float mu  = r1[0] * (1.f / 1024.f);
    float var = r2[0] * (1.f / 1024.f) - mu * mu;
    float rstd = rsqrtf(var + 1e-5f);
    const float4 gv = ((const float4*)g)[threadIdx.x];
    const float4 bv = ((const float4*)be)[threadIdx.x];
    __bf16* yr = y + (size_t)row * 1024 + threadIdx.x * 4;
    yr[0] = (__bf16)((v.x - mu) * rstd * gv.x + bv.x);
    yr[1] = (__bf16)((v.y - mu) * rstd * gv.y + bv.y);
    yr[2] = (__bf16)((v.z - mu) * rstd * gv.z + bv.z);
    yr[3] = (__bf16)((v.w - mu) * rstd * gv.w + bv.w);
}

// ---------------------------------------------------------------------------
// WMMA GEMM:  C = A[M][K](bf16) * Bt[N][K](bf16)^T  (+ bias, epilogue)
// 128x128 tile, 8 waves (4x2), wave = 32x64 = 2x4 16x16 tiles, K-step 32.
// Double-buffered LDS, register-staged global loads (one barrier / K-step).
// EPI: 0 = QKV store [b][h][t][d]    (bf16)
//      1 = V   store [b][h][d][t]    (bf16, transposed for PV)
//      2 = exact-erf GELU            (bf16)
//      3 = + resid                   (fp32)
// ---------------------------------------------------------------------------
__device__ __forceinline__ float gelu_exact(float x) {
    return 0.5f * x * (1.f + erff(x * 0.70710678118654752f));
}

template <int EPI>
__global__ __launch_bounds__(256) void gemm_wmma(
    const __bf16* __restrict__ A, const __bf16* __restrict__ Bt,
    const float* __restrict__ bias, const float* __restrict__ resid,
    float* __restrict__ Cf, __bf16* __restrict__ Cb,
    int M, int N, int K)
{
    __shared__ __align__(16) __bf16 As[2][128 * 40];
    __shared__ __align__(16) __bf16 Bs[2][128 * 40];

    const int lane = threadIdx.x & 31, w = threadIdx.x >> 5;
    const int wr = w >> 1, wc = w & 1;          // wave 4x2 grid
    const int m16 = lane & 15, hi = lane >> 4;  // fragment lane decode
    const int row0 = blockIdx.y * 128, col0 = blockIdx.x * 128;
    const int lr = threadIdx.x >> 1, lh = (threadIdx.x & 1) * 16;

    const __bf16* Aptr = A  + (size_t)(row0 + lr) * K + lh;
    const __bf16* Bptr = Bt + (size_t)(col0 + lr) * K + lh;
    const int sidx = lr * 40 + lh;

    v8f acc[2][4] = {};

    // prologue: stage first 128x32 tiles into LDS buffer 0
    uint4 ra0 = *(const uint4*)(Aptr);
    uint4 ra1 = *(const uint4*)(Aptr + 8);
    uint4 rb0 = *(const uint4*)(Bptr);
    uint4 rb1 = *(const uint4*)(Bptr + 8);
    *(uint4*)&As[0][sidx]     = ra0;
    *(uint4*)&As[0][sidx + 8] = ra1;
    *(uint4*)&Bs[0][sidx]     = rb0;
    *(uint4*)&Bs[0][sidx + 8] = rb1;
    __syncthreads();

    int cur = 0;
    for (int k0 = 0; k0 < K; k0 += 32) {
        const bool last = (k0 + 32 >= K);
        if (!last) {   // stage next tile into registers (overlaps with WMMAs)
            ra0 = *(const uint4*)(Aptr + k0 + 32);
            ra1 = *(const uint4*)(Aptr + k0 + 40);
            rb0 = *(const uint4*)(Bptr + k0 + 32);
            rb1 = *(const uint4*)(Bptr + k0 + 40);
        }

        v16bf af[2], bf[4];
        for (int rt = 0; rt < 2; ++rt) {   // A frag: 16x32, lane m, 16 contiguous K at hi*16
            const __bf16* p = &As[cur][(wr * 32 + rt * 16 + m16) * 40 + hi * 16];
            union { v16bf v; uint4 x[2]; } u;
            u.x[0] = *(const uint4*)p;
            u.x[1] = *(const uint4*)(p + 8);
            af[rt] = u.v;
        }
        for (int ct = 0; ct < 4; ++ct) {   // B frag: 32x16, lane n, 16 contiguous K at hi*16
            const __bf16* p = &Bs[cur][(wc * 64 + ct * 16 + m16) * 40 + hi * 16];
            union { v16bf v; uint4 x[2]; } u;
            u.x[0] = *(const uint4*)p;
            u.x[1] = *(const uint4*)(p + 8);
            bf[ct] = u.v;
        }
        for (int rt = 0; rt < 2; ++rt)
            for (int ct = 0; ct < 4; ++ct)
                acc[rt][ct] = __builtin_amdgcn_wmma_f32_16x16x32_bf16(
                    false, af[rt], false, bf[ct], (short)0, acc[rt][ct], false, false);

        if (!last) {   // commit staged registers to the other buffer
            *(uint4*)&As[cur ^ 1][sidx]     = ra0;
            *(uint4*)&As[cur ^ 1][sidx + 8] = ra1;
            *(uint4*)&Bs[cur ^ 1][sidx]     = rb0;
            *(uint4*)&Bs[cur ^ 1][sidx + 8] = rb1;
            __syncthreads();
            cur ^= 1;
        }
    }

    // epilogue: C/D layout -> row = j + hi*8, col = m16 per 16x16 tile
    for (int rt = 0; rt < 2; ++rt)
        for (int ct = 0; ct < 4; ++ct)
            for (int j = 0; j < 8; ++j) {
                int row = row0 + wr * 32 + rt * 16 + j + hi * 8;
                int col = col0 + wc * 64 + ct * 16 + m16;
                float v = acc[rt][ct][j] + bias[col];
                if (EPI == 0) {            // q/k -> [b][h][t][d]
                    int b = row >> 10, t = row & 1023, h = col >> 6, d = col & 63;
                    Cb[((size_t)((b * 16 + h) * 1024 + t)) * 64 + d] = (__bf16)v;
                } else if (EPI == 1) {     // v -> [b][h][d][t]
                    int b = row >> 10, t = row & 1023, h = col >> 6, d = col & 63;
                    Cb[((size_t)((b * 16 + h) * 64 + d)) * 1024 + t] = (__bf16)v;
                } else if (EPI == 2) {     // GELU -> bf16
                    Cb[(size_t)row * N + col] = (__bf16)gelu_exact(v);
                } else {                   // + residual -> fp32
                    Cf[(size_t)row * N + col] = v + resid[(size_t)row * N + col];
                }
            }
}

// ---------------------------------------------------------------------------
// Causal flash attention, per (b, h, 128-query block). 8 waves, wave-private
// 16-row query tile, 32-key chunks. WMMA for QK^T and PV; softmax via LDS.
// All K fragments of a chunk are loaded up-front; V fragments are loaded
// before the softmax so their latency hides under exp/LDS work.
// ---------------------------------------------------------------------------
__global__ __launch_bounds__(256) void attn_kernel(
    const __bf16* __restrict__ q, const __bf16* __restrict__ k,
    const __bf16* __restrict__ vT, __bf16* __restrict__ att)
{
    const int T = 1024, D = 64, H = 16;
    const int lane = threadIdx.x & 31, w = threadIdx.x >> 5;
    const int h = blockIdx.y, b = blockIdx.z;
    const int q0 = blockIdx.x * 128 + w * 16;
    const int m16 = lane & 15, hi = lane >> 4;

    const __bf16* qm = q  + (size_t)(b * H + h) * T * D;
    const __bf16* km = k  + (size_t)(b * H + h) * T * D;
    const __bf16* vm = vT + (size_t)(b * H + h) * D * T;

    __shared__ __align__(16) float  sS[8][16][32];
    __shared__ __align__(16) __bf16 sP[8][16][32];
    __shared__ float sM[8][16], sL[8][16], sC[8][16];

    // Q fragments (16x32 each) for d=[0,32) and [32,64)
    v16bf qf[2];
    for (int f = 0; f < 2; ++f) {
        union { v16bf v; uint4 x[2]; } u;
        const __bf16* p = qm + (size_t)(q0 + m16) * D + f * 32 + hi * 16;
        u.x[0] = *(const uint4*)p;
        u.x[1] = *(const uint4*)(p + 8);
        qf[f] = u.v;
    }
    v8f O[4] = {};
    if (lane < 16) { sM[w][lane] = -INFF; sL[w][lane] = 0.f; }

    const float scale = 0.125f;              // 1/sqrt(64)
    const int jmax = (q0 + 15) >> 5;
    for (int jc = 0; jc <= jmax; ++jc) {
        const int j0 = jc << 5;

        // ---- load ALL K fragments of this chunk (8 x b128, one clause) ----
        v16bf kf[2][2];
        for (int kt = 0; kt < 2; ++kt)
            for (int ds = 0; ds < 2; ++ds) {
                union { v16bf v; uint4 x[2]; } u;
                const __bf16* p = km + (size_t)(j0 + kt * 16 + m16) * D + ds * 32 + hi * 16;
                u.x[0] = *(const uint4*)p;
                u.x[1] = *(const uint4*)(p + 8);
                kf[kt][ds] = u.v;
            }

        // ---- S = Q K^T (two 16x16 key tiles, K over d=64) ----
        v8f s[2];
        for (int kt = 0; kt < 2; ++kt) {
            s[kt] = (v8f){};
            for (int ds = 0; ds < 2; ++ds)
                s[kt] = __builtin_amdgcn_wmma_f32_16x16x32_bf16(
                            false, qf[ds], false, kf[kt][ds], (short)0, s[kt], false, false);
        }

        // ---- load V fragments early (independent of softmax) ----
        v16bf vf[4];
        for (int t = 0; t < 4; ++t) {
            union { v16bf v; uint4 x[2]; } u;
            const __bf16* p = vm + (size_t)(t * 16 + m16) * T + j0 + hi * 16;
            u.x[0] = *(const uint4*)p;
            u.x[1] = *(const uint4*)(p + 8);
            vf[t] = u.v;
        }

        // ---- spill S to (wave-private) LDS ----
        for (int kt = 0; kt < 2; ++kt)
            for (int j = 0; j < 8; ++j)
                sS[w][j + hi * 8][kt * 16 + m16] = s[kt][j];
        asm volatile("s_wait_dscnt 0" ::: "memory");

        // ---- online softmax over the 32-key chunk (lanes 0..15 = rows) ----
        if (lane < 16) {
            const int qrow = q0 + lane;
            float mold = sM[w][lane];
            float mx = mold;
            for (int i = 0; i < 32; ++i)
                if (j0 + i <= qrow) mx = fmaxf(mx, sS[w][lane][i] * scale);
            float c = 1.f, sum = 0.f;
            if (mx > -INFF) {
                c = __expf(mold - mx);       // mold = -inf -> 0
                for (int i = 0; i < 32; ++i) {
                    float p = 0.f;
                    if (j0 + i <= qrow) p = __expf(sS[w][lane][i] * scale - mx);
                    sP[w][lane][i] = (__bf16)p;
                    sum += p;
                }
            } else {
                for (int i = 0; i < 32; ++i) sP[w][lane][i] = (__bf16)0.f;
            }
            sL[w][lane] = sL[w][lane] * c + sum;
            sM[w][lane] = mx;
            sC[w][lane] = c;
        }
        asm volatile("s_wait_dscnt 0" ::: "memory");

        // ---- rescale accumulator ----
        for (int j = 0; j < 8; ++j) {
            float c = sC[w][j + hi * 8];
            for (int t = 0; t < 4; ++t) O[t][j] *= c;
        }
        // ---- P fragment (16x32 A operand) ----
        union { v16bf v; uint4 x[2]; } up;
        {
            const __bf16* p = &sP[w][m16][hi * 16];
            up.x[0] = *(const uint4*)p;
            up.x[1] = *(const uint4*)(p + 8);
        }
        // ---- O += P @ V (4 d-tiles, B from transposed V, contiguous keys) ----
        for (int t = 0; t < 4; ++t)
            O[t] = __builtin_amdgcn_wmma_f32_16x16x32_bf16(
                       false, up.v, false, vf[t], (short)0, O[t], false, false);
    }
    // ---- finalize: divide by l, write [B,T,C] bf16 for the out-proj GEMM ----
    for (int j = 0; j < 8; ++j) {
        const int rl = j + hi * 8;
        const float inv = 1.f / sL[w][rl];
        const int row = q0 + rl;
        for (int t = 0; t < 4; ++t)
            att[((size_t)(b * T + row)) * 1024 + h * 64 + t * 16 + m16] =
                (__bf16)(O[t][j] * inv);
    }
}

// ---------------------------------------------------------------------------
// Launch: prep weights -> LN1 -> QKV -> attention -> proj+res -> LN2 -> MLP
// ---------------------------------------------------------------------------
extern "C" void kernel_launch(void* const* d_in, const int* in_sizes, int n_in,
                              void* d_out, int out_size, void* d_ws, size_t ws_size,
                              hipStream_t stream) {
    const float* x   = (const float*)d_in[0];
    const float* Wq  = (const float*)d_in[1];
    const float* bq  = (const float*)d_in[2];
    const float* Wk  = (const float*)d_in[3];
    const float* bk  = (const float*)d_in[4];
    const float* Wv  = (const float*)d_in[5];
    const float* bv  = (const float*)d_in[6];
    const float* Wp  = (const float*)d_in[7];
    const float* bp  = (const float*)d_in[8];
    const float* W1  = (const float*)d_in[9];
    const float* b1  = (const float*)d_in[10];
    const float* W2  = (const float*)d_in[11];
    const float* b2  = (const float*)d_in[12];
    const float* g1  = (const float*)d_in[13];
    const float* be1 = (const float*)d_in[14];
    const float* g2  = (const float*)d_in[15];
    const float* be2 = (const float*)d_in[16];
    float* out = (float*)d_out;

    const int M = 4096, C = 1024, F = 4096;
    char* ws = (char*)d_ws;
    size_t off = 0;
    auto take = [&](size_t bytes) -> char* {
        char* p = ws + off;
        off += (bytes + 255) & ~(size_t)255;
        return p;
    };
    __bf16* WqT  = (__bf16*)take((size_t)C * C * 2);
    __bf16* WkT  = (__bf16*)take((size_t)C * C * 2);
    __bf16* WvT  = (__bf16*)take((size_t)C * C * 2);
    __bf16* WpT  = (__bf16*)take((size_t)C * C * 2);
    __bf16* W1T  = (__bf16*)take((size_t)F * C * 2);
    __bf16* W2T  = (__bf16*)take((size_t)C * F * 2);
    __bf16* xln1 = (__bf16*)take((size_t)M * C * 2);
    __bf16* qb   = (__bf16*)take((size_t)M * C * 2);
    __bf16* kb_  = (__bf16*)take((size_t)M * C * 2);
    __bf16* vTb  = (__bf16*)take((size_t)M * C * 2);
    __bf16* attb = (__bf16*)take((size_t)M * C * 2);
    float*  x1   = (float*) take((size_t)M * C * 4);
    __bf16* xln2 = (__bf16*)take((size_t)M * C * 2);
    __bf16* hbuf = (__bf16*)take((size_t)M * F * 2);

    dim3 tb(32, 8);
    transpose_to_bf16<<<dim3(C / 32, C / 32), tb, 0, stream>>>(Wq, WqT, C, C);
    transpose_to_bf16<<<dim3(C / 32, C / 32), tb, 0, stream>>>(Wk, WkT, C, C);
    transpose_to_bf16<<<dim3(C / 32, C / 32), tb, 0, stream>>>(Wv, WvT, C, C);
    transpose_to_bf16<<<dim3(C / 32, C / 32), tb, 0, stream>>>(Wp, WpT, C, C);
    transpose_to_bf16<<<dim3(C / 32, F / 32), tb, 0, stream>>>(W1, W1T, C, F);
    transpose_to_bf16<<<dim3(F / 32, C / 32), tb, 0, stream>>>(W2, W2T, F, C);

    layernorm_bf16<<<M, 256, 0, stream>>>(x, g1, be1, xln1);

    dim3 gC(C / 128, M / 128);   // (8, 32)
    gemm_wmma<0><<<gC, 256, 0, stream>>>(xln1, WqT, bq, nullptr, nullptr, qb,  M, C, C);
    gemm_wmma<0><<<gC, 256, 0, stream>>>(xln1, WkT, bk, nullptr, nullptr, kb_, M, C, C);
    gemm_wmma<1><<<gC, 256, 0, stream>>>(xln1, WvT, bv, nullptr, nullptr, vTb, M, C, C);

    attn_kernel<<<dim3(8, 16, 4), 256, 0, stream>>>(qb, kb_, vTb, attb);

    gemm_wmma<3><<<gC, 256, 0, stream>>>(attb, WpT, bp, x, x1, nullptr, M, C, C);

    layernorm_bf16<<<M, 256, 0, stream>>>(x1, g2, be2, xln2);

    gemm_wmma<2><<<dim3(F / 128, M / 128), 256, 0, stream>>>(xln2, W1T, b1, nullptr, nullptr, hbuf, M, F, C);
    gemm_wmma<3><<<gC, 256, 0, stream>>>(hbuf, W2T, b2, x1, out, nullptr, M, C, F);
}
